// SimilarityBasedCrossEntropy_27650999452598
// MI455X (gfx1250) — matrix operand order repaired and checked by reading, not compiled, not run on hardware
//
#include <hip/hip_runtime.h>

#define N 8192
#define D 1024
#define SPLIT 2
#define CTS (N / 16 / SPLIT)          // 256 column tiles per split
#define ROWSTRIDE_B 2080              // LDS bytes per B row: 2048 + 2 pads of 16B (TDM pad)
#define TILEBYTES (16 * ROWSTRIDE_B)  // 33280 B per B tile buffer

typedef __attribute__((ext_vector_type(8)))  _Float16 v8h;
typedef __attribute__((ext_vector_type(16))) _Float16 v16h;
typedef __attribute__((ext_vector_type(8)))  float    v8f;
typedef __attribute__((ext_vector_type(4)))  unsigned v4u;
typedef __attribute__((ext_vector_type(4)))  int      v4i;
typedef __attribute__((ext_vector_type(8)))  int      v8i;

// ---------------- TDM helpers ----------------
__device__ __forceinline__ void wait_tensorcnt0() {
#if __has_builtin(__builtin_amdgcn_s_wait_tensorcnt)
    __builtin_amdgcn_s_wait_tensorcnt(0);
#else
    asm volatile("s_wait_tensorcnt 0x0" ::: "memory");
#endif
}

// DMA one 16-row x 1024-half B tile (row stride D halves in memory) into LDS at
// byte offset ldsOff, with TDM padding: 4 dwords pad every 256 dwords stored
// -> LDS layout offset(r,c) = r*2080 + 2*c + (c>=512 ? 16 : 0).
__device__ __forceinline__ void tdm_load_tile(const _Float16* gsrc, unsigned ldsOff) {
    union { v4u v; unsigned u[4]; } g0 = {};
    union { v8i v; int i[8]; } g1 = {};
    unsigned long long ga = (unsigned long long)(uintptr_t)gsrc;
    g0.u[0] = 1u;                                     // count=1, user descriptor
    g0.u[1] = ldsOff;                                 // lds_addr
    g0.u[2] = (unsigned)ga;                           // global_addr[31:0]
    g0.u[3] = (unsigned)(ga >> 32) | 0x80000000u;     // global_addr[56:32] | type=2
    // data_size=1 (2B) | pad_enable | pad_interval=7 (256 dw) | pad_amount=3 (4 dw)
    g1.i[0] = (1 << 16) | (1 << 20) | (7 << 22) | (3 << 25);
    g1.i[1] = (int)((1024u & 0xFFFFu) << 16);         // tensor_dim0 = 1024 (lo16)
    g1.i[2] = (int)((8192u & 0xFFFFu) << 16);         // dim0 hi16=0 | tensor_dim1 = 8192 (lo16)
    g1.i[3] = (int)(1024u << 16);                     // dim1 hi16=0 | tile_dim0 = 1024
    g1.i[4] = 16;                                     // tile_dim1 = 16, tile_dim2 = 0
    g1.i[5] = 1024;                                   // tensor_dim0_stride = 1024 (lo32)
    g1.i[6] = 0;                                      // stride hi16 | tensor_dim1_stride lo16
    g1.i[7] = 0;
    v4i z4 = {};
#if __has_builtin(__builtin_amdgcn_tensor_load_to_lds)
#if defined(__clang_major__) && (__clang_major__ >= 23)
    v8i z8 = {};
    __builtin_amdgcn_tensor_load_to_lds(g0.v, g1.v, z4, z4, z8, 0);
#else
    __builtin_amdgcn_tensor_load_to_lds(g0.v, g1.v, z4, z4, 0);
#endif
#else
    asm volatile("tensor_load_to_lds %0, %1, %2, %3"
                 :: "s"(g0.v), "s"(g1.v), "s"(z4), "s"(z4) : "memory");
#endif
}

// B fragment kk (K = kk*32 .. kk*32+31) for this lane from the padded LDS tile.
__device__ __forceinline__ v16h loadB(const char* curB, int kk) {
    const int off = kk * 64 + ((kk >= 16) ? 16 : 0);  // TDM mid-row pad at c>=512
    v8h blo = *(const v8h*)(curB + off);
    v8h bhi = *(const v8h*)(curB + off + 16);
    return __builtin_shufflevector(blo, bhi,
              0,1,2,3,4,5,6,7,8,9,10,11,12,13,14,15);
}

// Pin the tile-body schedule: 8 DS reads ahead, then [1 WMMA + 2 DS reads]*28,
// then the 4 drain WMMAs. Keeps 4 B fragments in flight (wait_dscnt <= 6).
__device__ __forceinline__ void pin_pipeline_schedule() {
#if __has_builtin(__builtin_amdgcn_sched_group_barrier)
    __builtin_amdgcn_sched_group_barrier(0x100, 8, 0);    // 8 DS reads (4 frags)
    #pragma unroll
    for (int i = 0; i < 28; ++i) {
        __builtin_amdgcn_sched_group_barrier(0x008, 1, 0); // 1 WMMA
        __builtin_amdgcn_sched_group_barrier(0x100, 2, 0); // 2 DS reads (next frag)
    }
    __builtin_amdgcn_sched_group_barrier(0x008, 4, 0);    // drain WMMAs
#endif
}

// ---------------- kernel 1: row L2-normalize fp32 -> f16 ----------------
__global__ void rownorm_f16(const float* __restrict__ in, _Float16* __restrict__ out) {
    __shared__ float red[256];
    const int row = blockIdx.x;
    const int tid = threadIdx.x;
    const float* r = in + (size_t)row * D;
    float ss = 0.f;
    for (int j = tid; j < D; j += 256) { float v = r[j]; ss += v * v; }
    red[tid] = ss;
    __syncthreads();
    for (int s = 128; s > 0; s >>= 1) {
        if (tid < s) red[tid] += red[tid + s];
        __syncthreads();
    }
    const float inv = 1.0f / fmaxf(sqrtf(red[0]), 1e-8f);   // torch eps clamp
    _Float16* o = out + (size_t)row * D;
    for (int j = tid; j < D; j += 256) o[j] = (_Float16)(r[j] * inv);
}

// ---------------- kernel 2: fused GEMM + row sum-exp ----------------
// 4 waves/WG, each wave owns 16 rows (A register-resident: 32 x v16h).
// B tiles staged by the Tensor Data Mover into double-buffered LDS shared by
// all 4 waves; DMA of tile t+1 overlaps compute of tile t. B fragments use a
// depth-4 rotating register buffer, enforced via sched_group_barrier.
__global__ void __launch_bounds__(128)
clip_lse(const _Float16* __restrict__ Ah, const _Float16* __restrict__ Bh,
         float* __restrict__ sumPart, float* __restrict__ diag) {
    extern __shared__ char smem[];                      // [2*TILEBYTES | red | diag]
    float* red = (float*)(smem + 2 * TILEBYTES);        // [4][16][16]
    float* dg  = (float*)(smem + 2 * TILEBYTES + 4096); // [4][16]

    const int tid    = threadIdx.x;
    const int wave   = tid >> 5;
    const int lane   = tid & 31;
    const int laneLo = lane & 15;
    const int hi     = lane >> 4;

    const int split   = blockIdx.x & (SPLIT - 1);
    const int rowBase = (blockIdx.x >> 1) * 64 + wave * 16;
    const int ct0     = split * CTS;

    // ---- A fragments straight from global into registers ----
    v16h afrag[32];
    {
        const _Float16* Ap = Ah + (size_t)(rowBase + laneLo) * D + hi * 16;
        #pragma unroll
        for (int kk = 0; kk < 32; ++kk) {
            v8h lo  = *(const v8h*)(Ap + kk * 32);
            v8h hi8 = *(const v8h*)(Ap + kk * 32 + 8);
            afrag[kk] = __builtin_shufflevector(lo, hi8,
                          0,1,2,3,4,5,6,7,8,9,10,11,12,13,14,15);
        }
    }

    // ---- prologue: DMA first tile into buffer 0 ----
    if (wave == 0) {
        tdm_load_tile(Bh + (size_t)(ct0 * 16) * D, 0u);
        wait_tensorcnt0();
    }
    __syncthreads();

    float accExp[8];
    #pragma unroll
    for (int i = 0; i < 8; ++i) accExp[i] = 0.f;
    const float K2 = 14.4269504088896340736f;           // log2(e)/T, T = 0.1

    for (int t = 0; t < CTS; ++t) {
        const int ct = ct0 + t;
        if (wave == 0 && t + 1 < CTS)                   // prefetch next tile via TDM
            tdm_load_tile(Bh + (size_t)((ct + 1) * 16) * D,
                          (unsigned)(((t + 1) & 1) * TILEBYTES));

        const char* curB = smem + (t & 1) * TILEBYTES + laneLo * ROWSTRIDE_B + hi * 32;

        // depth-4 software pipeline over the 32 K-steps
        v16h bb[4];
        #pragma unroll
        for (int p = 0; p < 4; ++p) bb[p] = loadB(curB, p);

        v8f c = {};
        #pragma unroll
        for (int kk = 0; kk < 32; ++kk) {
            c = __builtin_amdgcn_wmma_f32_16x16x32_f16(
                    false, afrag[kk], false, bb[kk & 3], (short)0, c, false, false);
            if (kk + 4 < 32) bb[kk & 3] = loadB(curB, kk + 4);
        }
        pin_pipeline_schedule();

        // c[r] = dot(m, n), m = r + 8*hi, n = laneLo; logits bounded: no max needed
        #pragma unroll
        for (int r = 0; r < 8; ++r)
            accExp[r] += exp2f(c[r] * K2);

        if (ct * 16 == rowBase) {                       // diagonal tile of this strip
            #pragma unroll
            for (int r = 0; r < 8; ++r) {
                int m = r + hi * 8;
                if (laneLo == m) dg[wave * 16 + m] = c[r];
            }
        }

        if (wave == 0) wait_tensorcnt0();               // next tile landed
        __syncthreads();                                // all done with cur, next ready
    }

    // ---- per-row reduction across the 16 lanes of each half-wave ----
    #pragma unroll
    for (int r = 0; r < 8; ++r)
        red[(wave * 16 + r + hi * 8) * 16 + laneLo] = accExp[r];
    __syncthreads();

    if (lane < 16) {
        float s = 0.f;
        #pragma unroll
        for (int j = 0; j < 16; ++j) s += red[(wave * 16 + lane) * 16 + j];
        const int row = rowBase + lane;
        sumPart[split * N + row] = s;
        const int rt = rowBase >> 4;
        if (rt >= ct0 && rt < ct0 + CTS)                // this split owns the diagonal
            diag[row] = dg[wave * 16 + lane];
    }
}

// ---------------- kernel 3: combine splits, mean loss ----------------
__global__ void reduce_mean(const float* __restrict__ sumPart,
                            const float* __restrict__ diag,
                            float* __restrict__ out) {
    __shared__ float red[256];
    const int tid = threadIdx.x;
    float acc = 0.f;
    for (int i = tid; i < N; i += 256) {
        float s = sumPart[i] + sumPart[N + i];
        acc += logf(s) - 10.0f * diag[i];               // lse_i - logits[i,i]
    }
    red[tid] = acc;
    __syncthreads();
    for (int s = 128; s > 0; s >>= 1) {
        if (tid < s) red[tid] += red[tid + s];
        __syncthreads();
    }
    if (tid == 0) out[0] = red[0] * (1.0f / (float)N);
}

extern "C" void kernel_launch(void* const* d_in, const int* in_sizes, int n_in,
                              void* d_out, int out_size, void* d_ws, size_t ws_size,
                              hipStream_t stream) {
    (void)in_sizes; (void)n_in; (void)out_size; (void)ws_size;
    const float* img = (const float*)d_in[0];
    const float* sen = (const float*)d_in[1];

    // ws: Ah[N*D] f16 | Bh[N*D] f16 | sumPart[2*N] f32 | diag[N] f32  (~32.1 MB)
    _Float16* Ah = (_Float16*)d_ws;
    _Float16* Bh = Ah + (size_t)N * D;
    float* sumPart = (float*)(Bh + (size_t)N * D);
    float* diag = sumPart + 2 * N;

    rownorm_f16<<<N, 256, 0, stream>>>(img, Ah);
    rownorm_f16<<<N, 256, 0, stream>>>(sen, Bh);
    const size_t smemBytes = 2 * TILEBYTES + 4096 + 256;   // B dbl-buf + red + diag
    clip_lse<<<(N / 64) * SPLIT, 128, smemBytes, stream>>>(Ah, Bh, sumPart, diag);
    reduce_mean<<<1, 256, 0, stream>>>(sumPart, diag, (float*)d_out);
}
